// StagedCounter_670014898339
// MI455X (gfx1250) — matrix operand order repaired and checked by reading, not compiled, not run on hardware
//
#include <hip/hip_runtime.h>

typedef __attribute__((ext_vector_type(16))) _Float16 v16h;
typedef __attribute__((ext_vector_type(8)))  float    v8f;

#define HH 96
#define WW 96
#define CS 4
#define DH 128      // adder hidden dim
#define MAXC 16384  // max chunk steps (schedule capacity)

// ---------------- init: zero counters ----------------
__global__ void k_init(unsigned int* rowTotal, unsigned int* rowPos, unsigned int* nChunks) {
    int t = threadIdx.x;
    if (t < HH) { rowTotal[t] = 0u; rowPos[t] = 0u; }
    if (t == 0) *nChunks = 0u;
}

// ---------------- stage 1: per-(b,r) stable compaction -> 96-bit positivity mask ----------------
__global__ void k_compact(const float* __restrict__ grid, const int* __restrict__ mask,
                          unsigned long long* __restrict__ bits,
                          unsigned int* rowTotal, unsigned int* rowPos, int B) {
    int idx = blockIdx.x * blockDim.x + threadIdx.x;
    if (idx >= B * HH) return;
    int b = idx / HH, r = idx - b * HH;
    const int base = (b * HH + r) * WW;
    unsigned long long lo = 0ull, hi = 0ull;
    int cnt = 0, pos = 0;
    for (int w = 0; w < WW; ++w) {
        if (mask[base + w] > 0) {
            float v = grid[base + w];
            if (v != 0.0f) {
                if (cnt < 64) lo |= (1ull << cnt); else hi |= (1ull << (cnt - 64));
                ++pos;
            }
            ++cnt;   // stable compaction order == increasing column
        }
    }
    bits[2 * idx] = lo;
    bits[2 * idx + 1] = hi;
    if (cnt) atomicAdd(&rowTotal[r], (unsigned)cnt);
    if (pos) atomicAdd(&rowPos[r], (unsigned)pos);
}

// ---------------- stage 2: chunk schedule + 16-entry subitize LUT ----------------
__global__ void k_sched(const unsigned int* rowTotal, const unsigned int* rowPos,
                        unsigned int* nChunks, unsigned int* rowIds, unsigned int* startsA,
                        const float* sew1, const float* seb1, const float* sew2, const float* seb2,
                        const float* scw1, const float* scb1, const float* scw2, const float* scb2,
                        float* lut) {
    int t = threadIdx.x;
    if (t == 0) {
        unsigned k = 0;
        for (int r = 0; r < HH; ++r) {
            unsigned n = rowTotal[r];
            if (n > 0u && rowPos[r] > 0u) {
                unsigned nc = (n + CS - 1) / CS;
                for (unsigned c = 0; c < nc && k < MAXC; ++c) { rowIds[k] = (unsigned)r; startsA[k] = c * CS; ++k; }
            }
        }
        *nChunks = k;
    }
    if (t < 16) {
        // subitize(pattern): feat=[bit, i/CS] -> relu MLP(2->64->64) -> mean over CS -> relu MLP(64->32->1) -> round
        float pooled[64];
        for (int j = 0; j < 64; ++j) pooled[j] = 0.0f;
        for (int i = 0; i < CS; ++i) {
            float f0 = ((t >> i) & 1) ? 1.0f : 0.0f;
            float f1 = (float)i / (float)CS;
            float e1[64];
            for (int j = 0; j < 64; ++j) {
                float s = f0 * sew1[j] + f1 * sew1[64 + j] + seb1[j];
                e1[j] = s > 0.0f ? s : 0.0f;
            }
            for (int j = 0; j < 64; ++j) {
                float s = seb2[j];
                for (int k2 = 0; k2 < 64; ++k2) s += e1[k2] * sew2[k2 * 64 + j];
                s = s > 0.0f ? s : 0.0f;
                pooled[j] += s * (1.0f / (float)CS);
            }
        }
        float s1[32];
        for (int j = 0; j < 32; ++j) {
            float s = scb1[j];
            for (int k2 = 0; k2 < 64; ++k2) s += pooled[k2] * scw1[k2 * 32 + j];
            s1[j] = s > 0.0f ? s : 0.0f;
        }
        float o = scb2[0];
        for (int k2 = 0; k2 < 32; ++k2) o += s1[k2] * scw2[k2];
        o = o > 0.0f ? o : 0.0f;
        lut[t] = rintf(o);
    }
}

// ---------------- stage 3: persistent per-wave scan with WMMA adder ----------------
// One wave (32 lanes) owns 16 batch rows; loops all chunk steps.
// Layer1 (K=2): packed-f16 FMA directly in WMMA A-operand layout (weights pre-packed per lane).
// Layer2 (128x128): v_wmma_f32_16x16x32_f16, 8 N-tiles x 4 K-tiles, bias folded into C.
// Layer3 (128->1): fused per-tile, cross-lane shfl_xor reduction.
__global__ void __launch_bounds__(32) k_scan(
    const unsigned long long* __restrict__ bits, const unsigned int* __restrict__ nChunksP,
    const unsigned int* __restrict__ rowIds, const unsigned int* __restrict__ startsA,
    const float* __restrict__ lutG,
    const float* __restrict__ adw1, const float* __restrict__ adb1,
    const float* __restrict__ adw2, const float* __restrict__ adb2,
    const float* __restrict__ adw3, const float* __restrict__ adb3,
    float* __restrict__ out, int B) {

    __shared__ v16h  lw2[4][8][32];            // w2 pre-packed in WMMA B-operand layout (f16), 32KB
    __shared__ float lw3[DH], lb2[DH];
    __shared__ float llut[16];

    const int L = threadIdx.x;
    const int half = L >> 4;        // 0: lanes 0-15, 1: lanes 16-31
    const int li = L & 15;

    for (int i = L; i < DH; i += 32) {
        lw3[i] = adw3[i];
        lb2[i] = adb2[i];
    }
    if (L < 16) llut[L] = lutG[L];

    // Pack w2 into B-operand layout: tile(t,n), lane L -> col = 16n+li, elem h -> K = 32t + (h<8 ? h : h+8) + 8*half
    for (int t = 0; t < 4; ++t)
        for (int n = 0; n < 8; ++n) {
            v16h v;
            const int col = n * 16 + li;
#pragma unroll
            for (int h = 0; h < 16; ++h) {
                const int K = t * 32 + (h < 8 ? h : h + 8) + (half ? 8 : 0);
                v[h] = (_Float16)adw2[K * DH + col];
            }
            lw2[t][n][L] = v;
        }

    // Pre-pack layer-1 weights as f16 in the A-operand ordering this lane needs.
    // A layout: lanes 0-15 row M=li: h<8 -> K=32t+h, h>=8 -> K=32t+h+8; lanes 16-31 same row, +8.
    v16h w10p[4], w11p[4], b1p[4];
#pragma unroll
    for (int t = 0; t < 4; ++t) {
#pragma unroll
        for (int h = 0; h < 16; ++h) {
            const int K = t * 32 + (h < 8 ? h : h + 8) + (half ? 8 : 0);
            w10p[t][h] = (_Float16)adw1[K];        // w1 row 0 (total feature)
            w11p[t][h] = (_Float16)adw1[DH + K];   // w1 row 1 (cnt feature)
            b1p[t][h]  = (_Float16)adb1[K];
        }
    }
    __syncthreads();

    const float b3 = adb3[0];
    const unsigned nC = *nChunksP;
    const int brow = blockIdx.x * 16 + li;  // this lane's batch row (duplicated across halves)
    float total = 0.0f;
    const v16h zeroh = {};

    for (unsigned s = 0; s < nC; ++s) {
        const unsigned r  = rowIds[s];
        const unsigned st = startsA[s];

        // 4-bit positivity pattern of this chunk for our batch row
        const unsigned long long lo = bits[((size_t)brow * HH + r) * 2];
        const unsigned long long hi = bits[((size_t)brow * HH + r) * 2 + 1];
        unsigned pat;
        if (st >= 128u) pat = 0u;
        else if (st >= 64u) pat = (unsigned)((hi >> (st - 64u)) & 0xFull);
        else {
            unsigned long long x = lo >> st;
            if (st) x |= hi << (64u - st);
            pat = (unsigned)(x & 0xFull);
        }
        const float cnt = llut[pat];

        const _Float16 x0h = (_Float16)(total * (1.0f / 50.0f));
        const _Float16 x1h = (_Float16)(cnt   * (1.0f / 50.0f));

        // Layer 1 (K=2) in packed f16, already in A-operand layout, with relu.
        v16h a[4];
#pragma unroll
        for (int t = 0; t < 4; ++t) {
            v16h sacc = b1p[t];
            sacc += w10p[t] * x0h;
            sacc += w11p[t] * x1h;
            a[t] = __builtin_elementwise_max(sacc, zeroh);
        }

        // Layer 2: h2 = relu(h1 @ w2 + b2) via WMMA; fuse layer 3 partials (h2 . w3).
        float p[8];
#pragma unroll
        for (int i = 0; i < 8; ++i) p[i] = 0.0f;

#pragma unroll
        for (int n = 0; n < 8; ++n) {
            const float bias = lb2[n * 16 + li];
            v8f acc;
#pragma unroll
            for (int i = 0; i < 8; ++i) acc[i] = bias;
#pragma unroll
            for (int t = 0; t < 4; ++t) {
                acc = __builtin_amdgcn_wmma_f32_16x16x32_f16(
                    /*neg_a=*/false, a[t], /*neg_b=*/false, lw2[t][n][L],
                    /*c_mod=*/(short)0, acc, /*reuse_a=*/false, /*reuse_b=*/false);
            }
            const float w3c = lw3[n * 16 + li];
#pragma unroll
            for (int i = 0; i < 8; ++i) {
                float h2 = acc[i];
                h2 = h2 > 0.0f ? h2 : 0.0f;
                p[i] += h2 * w3c;   // acc VGPR i == row (i + 8*half), col = 16n+li
            }
        }

        // Reduce the 16 column-lanes within each half-wave: p[i] -> full row dot
#pragma unroll
        for (int m = 1; m < 16; m <<= 1) {
#pragma unroll
            for (int i = 0; i < 8; ++i) p[i] += __shfl_xor(p[i], m, 32);
        }

        // Pick this lane's own row (li): rows 0-7 live in half 0, rows 8-15 in half 1.
        float v  = p[li & 7];
        float vo = __shfl_xor(v, 16, 32);
        const bool haveLocal = (half == ((li >> 3) & 1));
        const float chosen = haveLocal ? v : vo;

        total = rintf((chosen + b3) * 50.0f);   // round(adder(total, cnt))
    }

    if (half == 0 && brow < B) out[brow] = total;
}

extern "C" void kernel_launch(void* const* d_in, const int* in_sizes, int n_in,
                              void* d_out, int out_size, void* d_ws, size_t ws_size,
                              hipStream_t stream) {
    const float* grid = (const float*)d_in[0];
    const int*   mask = (const int*)d_in[1];
    const float* sew1 = (const float*)d_in[2];  const float* seb1 = (const float*)d_in[3];
    const float* sew2 = (const float*)d_in[4];  const float* seb2 = (const float*)d_in[5];
    const float* scw1 = (const float*)d_in[6];  const float* scb1 = (const float*)d_in[7];
    const float* scw2 = (const float*)d_in[8];  const float* scb2 = (const float*)d_in[9];
    const float* adw1 = (const float*)d_in[10]; const float* adb1 = (const float*)d_in[11];
    const float* adw2 = (const float*)d_in[12]; const float* adb2 = (const float*)d_in[13];
    const float* adw3 = (const float*)d_in[14]; const float* adb3 = (const float*)d_in[15];
    // d_in[16] = chunk_size (==4, baked into the 16-entry LUT)

    const int B = in_sizes[0] / (HH * WW);

    char* ws = (char*)d_ws;
    size_t off = 0;
    auto alloc = [&](size_t bytes) -> void* {
        off = (off + 255) & ~(size_t)255;
        void* p = (void*)(ws + off);
        off += bytes;
        return p;
    };
    unsigned long long* bits  = (unsigned long long*)alloc((size_t)B * HH * 16);
    unsigned int* rowTotal    = (unsigned int*)alloc(HH * 4);
    unsigned int* rowPos      = (unsigned int*)alloc(HH * 4);
    unsigned int* nChunks     = (unsigned int*)alloc(4);
    unsigned int* rowIds      = (unsigned int*)alloc((size_t)MAXC * 4);
    unsigned int* startsA     = (unsigned int*)alloc((size_t)MAXC * 4);
    float*        lut         = (float*)alloc(16 * 4);

    k_init<<<1, 128, 0, stream>>>(rowTotal, rowPos, nChunks);

    const int n = B * HH;
    k_compact<<<(n + 255) / 256, 256, 0, stream>>>(grid, mask, bits, rowTotal, rowPos, B);

    k_sched<<<1, 64, 0, stream>>>(rowTotal, rowPos, nChunks, rowIds, startsA,
                                  sew1, seb1, sew2, seb2, scw1, scb1, scw2, scb2, lut);

    k_scan<<<B / 16, 32, 0, stream>>>(bits, nChunks, rowIds, startsA, lut,
                                      adw1, adb1, adw2, adb2, adw3, adb3,
                                      (float*)d_out, B);
}
